// SSAA_47914655154294
// MI455X (gfx1250) — matrix operand order repaired and checked by reading, not compiled, not run on hardware
//
#include <hip/hip_runtime.h>
#include <hip/hip_bf16.h>

// ---------------------------------------------------------------------------
// MI455X (gfx1250): bf16 WMMA pipeline, fragment-friendly bf16 operand
// layouts (A row-major [M][K], B transposed [N][K] -> every fragment is two
// global_load_b128 per lane), with 64x64 register blocking (4x4 fragment
// tiles, 16 WMMAs per K-step) for 32 flops/byte of fragment traffic.
// ---------------------------------------------------------------------------

typedef __attribute__((ext_vector_type(16))) __bf16 bf16x16;
typedef __attribute__((ext_vector_type(8)))  float  floatx8;
typedef __attribute__((ext_vector_type(4)))  unsigned int uint4v;

#define CCH   256
#define NHEAD 8
#define HD    32
#define NQ    4096   // 64*64
#define NKV   256    // 16*16
#define BATCH 8
#define ATT_SCALE 0.17677669529663687f  // 32^-0.5

// ---- bf16 scalar helper (converters / epilogues only) ---------------------
static __device__ inline unsigned short f2bfbits(float f) {
  unsigned int u = __float_as_uint(f);
  u += 0x7FFFu + ((u >> 16) & 1u);           // round-to-nearest-even
  return (unsigned short)(u >> 16);
}

// ---- packed fragment loads (CDNA5 ISA 7.12.2 layouts, wave32) -------------
static __device__ inline bf16x16 load_a_frag(const unsigned short* A, int lda,
                                             int m0, int k0, int lane) {
  const int half = (lane >> 4) & 1;
  const unsigned short* p = A + (size_t)(m0 + (lane & 15)) * lda + k0 + half * 8;
  union { bf16x16 v; uint4v u[2]; } f;
  f.u[0] = *(const uint4v*)(p);
  f.u[1] = *(const uint4v*)(p + 16);
  return f.v;
}
static __device__ inline bf16x16 load_b_frag(const unsigned short* Bt, int ldb,
                                             int n0, int k0, int lane) {
  const int half = (lane >> 4) & 1;
  const unsigned short* p = Bt + (size_t)(n0 + (lane & 15)) * ldb + k0 + half * 8;
  union { bf16x16 v; uint4v u[2]; } f;
  f.u[0] = *(const uint4v*)(p);
  f.u[1] = *(const uint4v*)(p + 16);
  return f.v;
}
static __device__ inline floatx8 wmma_bf16(bf16x16 a, bf16x16 b, floatx8 c) {
  return __builtin_amdgcn_wmma_f32_16x16x32_bf16(false, a, false, b,
                                                 (short)0, c, false, false);
}
// D/C f32 16x16: m = half*8 + r, n = lane & 15.

// ---- 64x64 register-blocked GEMM core (4x4 tiles, 16 WMMA per K-step) -----
template <int KT>
static __device__ inline void gemm64x64(const unsigned short* __restrict__ A, int lda, int m0,
                                        const unsigned short* __restrict__ Bt, int ldb, int n0,
                                        int lane, floatx8 acc[4][4]) {
  for (int k0 = 0; k0 < KT; k0 += 32) {
    bf16x16 af[4], bfr[4];
#pragma unroll
    for (int i = 0; i < 4; ++i) af[i] = load_a_frag(A, lda, m0 + 16 * i, k0, lane);
#pragma unroll
    for (int j = 0; j < 4; ++j) bfr[j] = load_b_frag(Bt, ldb, n0 + 16 * j, k0, lane);
#pragma unroll
    for (int i = 0; i < 4; ++i)
#pragma unroll
      for (int j = 0; j < 4; ++j)
        acc[i][j] = wmma_bf16(af[i], bfr[j], acc[i][j]);
  }
}

// ---------------------------------------------------------------------------
// Converters (bandwidth-trivial pre-passes)
// ---------------------------------------------------------------------------
__global__ void __launch_bounds__(256)
k_cvt(const float* __restrict__ src, unsigned short* __restrict__ dst, int count) {
  const int i = blockIdx.x * blockDim.x + threadIdx.x;
  if (i < count) dst[i] = f2bfbits(src[i]);
}

// xT[b][n][c] = x[b][c][n]   (B operand for q projection)
__global__ void __launch_bounds__(256)
k_build_xT(const float* __restrict__ x, unsigned short* __restrict__ xT) {
  const int i = blockIdx.x * blockDim.x + threadIdx.x;   // 8*4096*256
  const int c = i & 255, n = (i >> 8) & 4095, b = i >> 20;
  xT[i] = f2bfbits(x[((size_t)b * CCH + c) * NQ + n]);
}

// xim[b][p][k], k = ci*16 + ky*4 + kx  (B operand for stride-4 conv im2col)
__global__ void __launch_bounds__(256)
k_build_xim(const float* __restrict__ x, unsigned short* __restrict__ xim) {
  const int i = blockIdx.x * blockDim.x + threadIdx.x;   // 8*256*4096
  const int k = i & 4095, p = (i >> 12) & 255, b = i >> 20;
  const int ci = k >> 4, rr = k & 15, ky = rr >> 2, kx = rr & 3;
  const int py = p >> 4, px = p & 15;
  xim[i] = f2bfbits(x[((size_t)(b * CCH + ci) * 64 + py * 4 + ky) * 64 + px * 4 + kx]);
}

// ---------------------------------------------------------------------------
// Kernel: q projection.  M=512, N=4096, K=256 per batch; 64x64 per wave.
// Output: qbuf[(b*16 + (c2>>5))][q][d] bf16 (attention A-operand layout).
// ---------------------------------------------------------------------------
__global__ void __launch_bounds__(256)
k_qproj(const unsigned short* __restrict__ xT, const unsigned short* __restrict__ wqb,
        const float* __restrict__ bq, unsigned short* __restrict__ qbuf) {
  const int wave = (blockIdx.x * blockDim.x + threadIdx.x) >> 5;
  const int lane = threadIdx.x & 31;
  const int nb = wave & 63, mb = (wave >> 6) & 7, b = wave >> 9;
  const unsigned short* Bt = xT + (size_t)b * NQ * CCH;

  floatx8 acc[4][4];
  floatx8 z = {};
#pragma unroll
  for (int i = 0; i < 4; ++i)
#pragma unroll
    for (int j = 0; j < 4; ++j) acc[i][j] = z;

  gemm64x64<CCH>(wqb, CCH, mb * 64, Bt, CCH, nb * 64, lane, acc);

  const int half = lane >> 4, n = lane & 15;
#pragma unroll
  for (int i = 0; i < 4; ++i)
#pragma unroll
    for (int j = 0; j < 4; ++j)
#pragma unroll
      for (int r = 0; r < 8; ++r) {
        const int mg = mb * 64 + i * 16 + half * 8 + r;   // channel c2
        const int ng = nb * 64 + j * 16 + n;              // query
        qbuf[(size_t)(b * 16 + (mg >> 5)) * (NQ * HD) + (size_t)ng * HD + (mg & 31)] =
            f2bfbits(acc[i][j][r] + bq[mg]);
      }
}

// ---------------------------------------------------------------------------
// Kernel: stride-4 conv GEMM.  M=256, N=256, K=4096; 64x64 per wave.
// Outputs xr fp32 [b][c][p] and xrT bf16 [b][p][c].
// ---------------------------------------------------------------------------
__global__ void __launch_bounds__(256)
k_sconv(const unsigned short* __restrict__ xim, const unsigned short* __restrict__ wsb,
        const float* __restrict__ bs, float* __restrict__ xr,
        unsigned short* __restrict__ xrT) {
  const int wave = (blockIdx.x * blockDim.x + threadIdx.x) >> 5;
  const int lane = threadIdx.x & 31;
  const int nb = wave & 3, mb = (wave >> 2) & 3, b = wave >> 4;
  const unsigned short* Bt = xim + (size_t)b * NKV * 4096;

  floatx8 acc[4][4];
  floatx8 z = {};
#pragma unroll
  for (int i = 0; i < 4; ++i)
#pragma unroll
    for (int j = 0; j < 4; ++j) acc[i][j] = z;

  gemm64x64<4096>(wsb, 4096, mb * 64, Bt, 4096, nb * 64, lane, acc);

  const int half = lane >> 4, n = lane & 15;
#pragma unroll
  for (int i = 0; i < 4; ++i)
#pragma unroll
    for (int j = 0; j < 4; ++j)
#pragma unroll
      for (int r = 0; r < 8; ++r) {
        const int mg = mb * 64 + i * 16 + half * 8 + r;   // out channel
        const int ng = nb * 64 + j * 16 + n;              // spatial
        const float v = acc[i][j][r] + bs[mg];
        xr[((size_t)b * CCH + mg) * NKV + ng] = v;
        xrT[((size_t)b * NKV + ng) * CCH + mg] = f2bfbits(v);
      }
}

// ---------------------------------------------------------------------------
// Kernel: depthwise 3x3 (pad 1) -> transposed bf16 [b][p][c], both branches.
// ---------------------------------------------------------------------------
__global__ void __launch_bounds__(256)
k_dw3x3(const float* __restrict__ xr,
        const float* __restrict__ wkd, const float* __restrict__ bkd,
        const float* __restrict__ wvd, const float* __restrict__ bvd,
        unsigned short* __restrict__ dwkT, unsigned short* __restrict__ dwvT) {
  const int idx = blockIdx.x * blockDim.x + threadIdx.x;   // c fastest
  if (idx >= BATCH * CCH * NKV) return;
  const int cch = idx & 255, p = (idx >> 8) & 255, b = idx >> 16;
  const int py = p >> 4, px = p & 15;
  const size_t base = (size_t)(b * CCH + cch) * NKV;
  float sk = bkd[cch], sv = bvd[cch];
#pragma unroll
  for (int dy = -1; dy <= 1; ++dy) {
#pragma unroll
    for (int dx = -1; dx <= 1; ++dx) {
      const int yy = py + dy, xx = px + dx;
      if (yy >= 0 && yy < 16 && xx >= 0 && xx < 16) {
        const float v = xr[base + yy * 16 + xx];
        const int wi = cch * 9 + (dy + 1) * 3 + (dx + 1);
        sk += wkd[wi] * v;
        sv += wvd[wi] * v;
      }
    }
  }
  const size_t o = (size_t)(b * NKV + p) * CCH + cch;
  dwkT[o] = f2bfbits(sk);
  dwvT[o] = f2bfbits(sv);
}

// ---------------------------------------------------------------------------
// Kernel: 1x1 KV projection GEMM.  M=256, N=256, K=256; 64x64 per wave.
// vmode=0 -> K layout [b*8+h][kv][32];  vmode=1 -> V layout [b*8+h][32][kv].
// ---------------------------------------------------------------------------
__global__ void __launch_bounds__(256)
k_kvproj(const unsigned short* __restrict__ srcT, const unsigned short* __restrict__ wb,
         const float* __restrict__ bias, unsigned short* __restrict__ dst, int vmode) {
  const int wave = (blockIdx.x * blockDim.x + threadIdx.x) >> 5;
  const int lane = threadIdx.x & 31;
  const int nb = wave & 3, mb = (wave >> 2) & 3, b = wave >> 4;
  const unsigned short* Bt = srcT + (size_t)b * NKV * CCH;

  floatx8 acc[4][4];
  floatx8 z = {};
#pragma unroll
  for (int i = 0; i < 4; ++i)
#pragma unroll
    for (int j = 0; j < 4; ++j) acc[i][j] = z;

  gemm64x64<CCH>(wb, CCH, mb * 64, Bt, CCH, nb * 64, lane, acc);

  const int half = lane >> 4, n = lane & 15;
#pragma unroll
  for (int i = 0; i < 4; ++i)
#pragma unroll
    for (int j = 0; j < 4; ++j)
#pragma unroll
      for (int r = 0; r < 8; ++r) {
        const int mg = mb * 64 + i * 16 + half * 8 + r;   // channel = h*32 + d
        const int ng = nb * 64 + j * 16 + n;              // kv position
        const unsigned short v = f2bfbits(acc[i][j][r] + bias[mg]);
        if (vmode == 0)
          dst[((size_t)(b * NHEAD + (mg >> 5)) * NKV + ng) * HD + (mg & 31)] = v;
        else
          dst[((size_t)(b * NHEAD + (mg >> 5)) * HD + (mg & 31)) * NKV + ng] = v;
      }
}

// ---------------------------------------------------------------------------
// Kernel: attention.  One wave owns a 16-query tile of one (b, branch, head).
// Q-fragment reused across all 16 score WMMAs; softmax staged in fp32 LDS;
// normalized P written bf16 to LDS; PV fed by ds_load_b128 (16 WMMAs).
// ---------------------------------------------------------------------------
__global__ void __launch_bounds__(64)
k_attn(const unsigned short* __restrict__ qbuf,
       const unsigned short* __restrict__ kA, const unsigned short* __restrict__ vA,
       const unsigned short* __restrict__ kE, const unsigned short* __restrict__ vE,
       unsigned short* __restrict__ ocT) {
  __shared__ float sbuf[2][16][NKV];                        // 32 KB
  __shared__ __align__(16) unsigned short pbuf[2][16][NKV]; // 16 KB
  const int lane = threadIdx.x & 31;
  const int wloc = threadIdx.x >> 5;
  const int wave = blockIdx.x * 2 + wloc;
  const int qt = wave & 255;
  const int h  = (wave >> 8) & 7;
  const int br = (wave >> 11) & 1;
  const int b  = wave >> 12;
  float (*S)[NKV] = sbuf[wloc];
  unsigned short (*P)[NKV] = pbuf[wloc];

  const size_t headoff = (size_t)(b * NHEAD + h) * NKV * HD;
  const unsigned short* Kt = (br ? kE : kA) + headoff;   // [kv][32]
  const unsigned short* Vt = (br ? vE : vA) + headoff;   // [32][kv]
  const unsigned short* Qt = qbuf + (size_t)(b * 16 + br * 8 + h) * (NQ * HD)
                                  + (size_t)qt * 16 * HD; // [q][32]

  const int half = lane >> 4, nlow = lane & 15;

  // ---- scores
  bf16x16 aq = load_a_frag(Qt, HD, 0, 0, lane);
  for (int kt = 0; kt < 16; ++kt) {
    bf16x16 bk = load_b_frag(Kt, HD, kt * 16, 0, lane);
    floatx8 c = {};
    c = wmma_bf16(aq, bk, c);
#pragma unroll
    for (int r = 0; r < 8; ++r)
      S[half * 8 + r][kt * 16 + nlow] = c[r] * ATT_SCALE;
  }

  // ---- softmax: lanes L and L+16 split row L&15 (DS in-order within wave)
  const int row = lane & 15;
  const int cs  = half * 128;
  float mx = -3.0e30f;
  for (int i = 0; i < 128; ++i) mx = fmaxf(mx, S[row][cs + i]);
  mx = fmaxf(mx, __shfl_xor(mx, 16, 32));
  float sum = 0.0f;
  for (int i = 0; i < 128; ++i) {
    const float e = __expf(S[row][cs + i] - mx);
    S[row][cs + i] = e;
    sum += e;
  }
  sum += __shfl_xor(sum, 16, 32);
  const float inv = 1.0f / sum;
  for (int i = 0; i < 128; ++i)
    P[row][cs + i] = f2bfbits(S[row][cs + i] * inv);

  // ---- O[m][d] = sum_kv P[m,kv] V[kv,d]  (two 16-wide d tiles)
  floatx8 o0 = {}, o1 = {};
  for (int kc = 0; kc < NKV; kc += 32) {
    bf16x16 ap = load_a_frag(&P[0][0], NKV, 0, kc, lane);
    bf16x16 b0 = load_b_frag(Vt, NKV, 0, kc, lane);
    o0 = wmma_bf16(ap, b0, o0);
    bf16x16 b1 = load_b_frag(Vt, NKV, 16, kc, lane);
    o1 = wmma_bf16(ap, b1, o1);
  }

  const int chbase = br * 256 + h * 32;
#pragma unroll
  for (int r = 0; r < 8; ++r) {
    const int m = half * 8 + r;                // query within tile
    const size_t rowoff = ((size_t)b * NQ + qt * 16 + m) * 512;
    ocT[rowoff + chbase + nlow]      = f2bfbits(o0[r]);
    ocT[rowoff + chbase + 16 + nlow] = f2bfbits(o1[r]);
  }
}

// ---------------------------------------------------------------------------
// Kernel: final 1x1 projection.  M=256, N=4096, K=512; 64x64 per wave.
// ---------------------------------------------------------------------------
__global__ void __launch_bounds__(256)
k_proj(const unsigned short* __restrict__ ocT, const unsigned short* __restrict__ wpb,
       const float* __restrict__ bproj, float* __restrict__ out) {
  const int wave = (blockIdx.x * blockDim.x + threadIdx.x) >> 5;
  const int lane = threadIdx.x & 31;
  const int nb = wave & 63, mb = (wave >> 6) & 3, b = wave >> 8;
  const unsigned short* Bt = ocT + (size_t)b * NQ * 512;

  floatx8 acc[4][4];
  floatx8 z = {};
#pragma unroll
  for (int i = 0; i < 4; ++i)
#pragma unroll
    for (int j = 0; j < 4; ++j) acc[i][j] = z;

  gemm64x64<512>(wpb, 512, mb * 64, Bt, 512, nb * 64, lane, acc);

  const int half = lane >> 4, n = lane & 15;
#pragma unroll
  for (int i = 0; i < 4; ++i)
#pragma unroll
    for (int j = 0; j < 4; ++j)
#pragma unroll
      for (int r = 0; r < 8; ++r) {
        const int mg = mb * 64 + i * 16 + half * 8 + r;
        const int ng = nb * 64 + j * 16 + n;
        out[((size_t)b * CCH + mg) * NQ + ng] = acc[i][j][r] + bproj[mg];
      }
}

// ---------------------------------------------------------------------------
extern "C" void kernel_launch(void* const* d_in, const int* in_sizes, int n_in,
                              void* d_out, int out_size, void* d_ws, size_t ws_size,
                              hipStream_t stream) {
  const float* x      = (const float*)d_in[0];
  const float* wq     = (const float*)d_in[1];
  const float* bq     = (const float*)d_in[2];
  const float* ws     = (const float*)d_in[3];
  const float* bs     = (const float*)d_in[4];
  const float* wak_d  = (const float*)d_in[5];
  const float* bak_d  = (const float*)d_in[6];
  const float* wak_p  = (const float*)d_in[7];
  const float* bak_p  = (const float*)d_in[8];
  const float* wav_d  = (const float*)d_in[9];
  const float* bav_d  = (const float*)d_in[10];
  const float* wav_p  = (const float*)d_in[11];
  const float* bav_p  = (const float*)d_in[12];
  const float* wek    = (const float*)d_in[13];
  const float* bek    = (const float*)d_in[14];
  const float* wev    = (const float*)d_in[15];
  const float* bev    = (const float*)d_in[16];
  const float* wproj  = (const float*)d_in[17];
  const float* bproj  = (const float*)d_in[18];
  float* out = (float*)d_out;

  // Workspace carve-up (~111 MB), all chunks 256B-aligned by construction.
  char* wsp = (char*)d_ws;
  auto take = [&](size_t bytes) { char* p = wsp; wsp += (bytes + 255) & ~(size_t)255; return p; };
  unsigned short* wq_bf   = (unsigned short*)take(512 * 256 * 2);
  unsigned short* ws_bf   = (unsigned short*)take(256 * 4096 * 2);
  unsigned short* wakp_bf = (unsigned short*)take(256 * 256 * 2);
  unsigned short* wavp_bf = (unsigned short*)take(256 * 256 * 2);
  unsigned short* wek_bf  = (unsigned short*)take(256 * 256 * 2);
  unsigned short* wev_bf  = (unsigned short*)take(256 * 256 * 2);
  unsigned short* wp_bf   = (unsigned short*)take(256 * 512 * 2);
  unsigned short* xT      = (unsigned short*)take((size_t)BATCH * NQ * CCH * 2);
  unsigned short* xim     = (unsigned short*)take((size_t)BATCH * NKV * 4096 * 2);
  float*          xr      = (float*)take((size_t)BATCH * CCH * NKV * 4);
  unsigned short* xrT     = (unsigned short*)take((size_t)BATCH * NKV * CCH * 2);
  unsigned short* dwkT    = (unsigned short*)take((size_t)BATCH * NKV * CCH * 2);
  unsigned short* dwvT    = (unsigned short*)take((size_t)BATCH * NKV * CCH * 2);
  const size_t kvt = (size_t)BATCH * NHEAD * NKV * HD;
  unsigned short* kA      = (unsigned short*)take(kvt * 2);
  unsigned short* vA      = (unsigned short*)take(kvt * 2);
  unsigned short* kE      = (unsigned short*)take(kvt * 2);
  unsigned short* vE      = (unsigned short*)take(kvt * 2);
  unsigned short* qbuf    = (unsigned short*)take((size_t)BATCH * 512 * NQ * 2);
  unsigned short* ocT     = (unsigned short*)take((size_t)BATCH * NQ * 512 * 2);

  // -- weight conversions (tiny)
  k_cvt<<<512, 256, 0, stream>>>(wq, wq_bf, 512 * 256);
  k_cvt<<<4096, 256, 0, stream>>>(ws, ws_bf, 256 * 4096);
  k_cvt<<<256, 256, 0, stream>>>(wak_p, wakp_bf, 256 * 256);
  k_cvt<<<256, 256, 0, stream>>>(wav_p, wavp_bf, 256 * 256);
  k_cvt<<<256, 256, 0, stream>>>(wek, wek_bf, 256 * 256);
  k_cvt<<<256, 256, 0, stream>>>(wev, wev_bf, 256 * 256);
  k_cvt<<<512, 256, 0, stream>>>(wproj, wp_bf, 256 * 512);
  // -- activation staging
  k_build_xT<<<32768, 256, 0, stream>>>(x, xT);
  k_build_xim<<<32768, 256, 0, stream>>>(x, xim);

  // -- GEMM pipeline (64x64 per wave)
  k_qproj<<<512, 256, 0, stream>>>(xT, wq_bf, bq, qbuf);          // 4096 waves
  k_sconv<<<16, 256, 0, stream>>>(xim, ws_bf, bs, xr, xrT);       // 128 waves
  k_dw3x3<<<2048, 256, 0, stream>>>(xr, wak_d, bak_d, wav_d, bav_d, dwkT, dwvT);
  k_kvproj<<<16, 256, 0, stream>>>(dwkT, wakp_bf, bak_p, kA, 0);
  k_kvproj<<<16, 256, 0, stream>>>(dwvT, wavp_bf, bav_p, vA, 1);
  k_kvproj<<<16, 256, 0, stream>>>(xrT,  wek_bf,  bek,  kE, 0);
  k_kvproj<<<16, 256, 0, stream>>>(xrT,  wev_bf,  bev,  vE, 1);
  k_attn<<<16384, 64, 0, stream>>>(qbuf, kA, vA, kE, vE, ocT);
  k_proj<<<1024, 256, 0, stream>>>(ocT, wp_bf, bproj, out);       // 8192 waves
}
